// Head_3212635537781
// MI455X (gfx1250) — matrix operand order repaired and checked by reading, not compiled, not run on hardware
//
#include <hip/hip_runtime.h>
#include <hip/hip_bf16.h>

// ---------------------------------------------------------------------------
// Single-head causal attention (B=8, T=2048, C=1024, H=64) for gfx1250.
// Flash-attention structure, bf16 WMMA (v_wmma_f32_16x16x32_bf16) everywhere.
// Attention blocks double-buffer q/V tiles in LDS via ASYNC global->LDS DMA
// (global_load_async_to_lds_b128 + s_wait_asynccnt), one barrier per chunk.
// ---------------------------------------------------------------------------

#define Bx   8
#define Tx   2048
#define Cx   1024
#define Hx   64

#define QS_STRIDE 72   // 144B rows: bank step 36 -> 16 disjoint lane windows
#define VS_STRIDE 56   // 112B rows: bank step 28 -> 16 disjoint lane windows
#define PB_STRIDE 56

typedef __attribute__((ext_vector_type(16))) __bf16        v16bf;
typedef __attribute__((ext_vector_type(8)))  float         v8f;
typedef __attribute__((ext_vector_type(4)))  float         f32x4;
typedef __attribute__((ext_vector_type(4)))  unsigned int  u32x4;

union Frag16 {            // 16 bf16 = 8 dwords = one WMMA A/B fragment
    u32x4        u[2];
    unsigned int w[8];
    v16bf        v;
};

__device__ __forceinline__ unsigned short f2bf_bits(float f) {
    unsigned u = __builtin_bit_cast(unsigned, f);
    u += 0x7FFFu + ((u >> 16) & 1u);          // round-to-nearest-even
    return (unsigned short)(u >> 16);
}

// Async 16B global->LDS copy (ASYNCcnt-tracked DMA; no VGPR round-trip).
// LDS aperture is 4GB-aligned, so low 32 bits of a generic LDS pointer are
// the wave-relative LDS byte address the instruction wants in VDST.
__device__ __forceinline__ void async_copy16(unsigned short* lds_dst,
                                             const unsigned short* gsrc) {
    unsigned lds_addr = (unsigned)(size_t)lds_dst;
    asm volatile("global_load_async_to_lds_b128 %0, %1, off"
                 :: "v"(lds_addr), "v"(gsrc) : "memory");
}

__device__ __forceinline__ void wait_asynccnt0() {
#if __has_builtin(__builtin_amdgcn_s_wait_asynccnt)
    __builtin_amdgcn_s_wait_asynccnt(0);
#else
    asm volatile("s_wait_asynccnt 0x0" ::: "memory");
#endif
}

// ---------------------------------------------------------------------------
// Kernel 1a: W[1024][64] fp32  ->  WT[64][1024] bf16  (x3 matrices)
// ---------------------------------------------------------------------------
__global__ void prep_w(const float* __restrict__ Wq,
                       const float* __restrict__ Wk,
                       const float* __restrict__ Wv,
                       unsigned short* __restrict__ wT) {
    int idx = blockIdx.x * blockDim.x + threadIdx.x;
    if (idx >= 3 * Hx * Cx) return;
    int m = idx >> 16;              // which matrix
    int o = idx & 65535;            // n*1024 + k
    int n = o >> 10;
    int k = o & (Cx - 1);
    const float* W = (m == 0) ? Wq : ((m == 1) ? Wk : Wv);
    wT[idx] = f2bf_bits(W[k * Hx + n]);
}

// ---------------------------------------------------------------------------
// Kernel 1b: x fp32 -> bf16 (bandwidth-bound; keeps conversion VALU out of
// the GEMM loop).  8 elements per thread: 2x b128 load, 1x b128 store.
// ---------------------------------------------------------------------------
__global__ __launch_bounds__(256) void prep_x(const float* __restrict__ x,
                                              unsigned short* __restrict__ xb) {
    size_t e = ((size_t)blockIdx.x * blockDim.x + threadIdx.x) * 8;
    f32x4 f0 = *(const f32x4*)(x + e);
    f32x4 f1 = *(const f32x4*)(x + e + 4);
    unsigned short h[8];
    #pragma unroll
    for (int i = 0; i < 4; ++i) { h[i] = f2bf_bits(f0[i]); h[4 + i] = f2bf_bits(f1[i]); }
    *(u32x4*)(xb + e) = *(const u32x4*)h;
}

// ---------------------------------------------------------------------------
// Kernel 2 (fast path): QKV projection from pre-converted bf16 x.
// One wave -> 16 rows x 64 cols of q, k AND v (12 WMMA accumulators).
//   qb, kb : bf16 [B*T][64]   (d-contiguous)
//   vtb    : bf16 [B][64][T]  (j-contiguous, i.e. v transposed)
// ---------------------------------------------------------------------------
__global__ __launch_bounds__(128) void qkv_proj_bf16(
        const unsigned short* __restrict__ xb,
        const unsigned short* __restrict__ wT,   // [3][64][1024] bf16
        unsigned short* __restrict__ qb,
        unsigned short* __restrict__ kb,
        unsigned short* __restrict__ vtb) {
    const int lane = threadIdx.x & 31;
    const int lo   = lane & 15;
    const int half = lane >> 4;
    const int wid  = blockIdx.x * (blockDim.x >> 5) + (threadIdx.x >> 5);
    const int rowbase = wid << 4;
    const int arow = rowbase + lo;

    const v8f vz = {0.f, 0.f, 0.f, 0.f, 0.f, 0.f, 0.f, 0.f};
    v8f acc[3][4];
    #pragma unroll
    for (int m = 0; m < 3; ++m)
        #pragma unroll
        for (int nt = 0; nt < 4; ++nt) acc[m][nt] = vz;

    for (int kc = 0; kc < Cx; kc += 32) {
        // A fragment (ISA 16-bit A layout): two b128 loads
        const unsigned short* xr = xb + (size_t)arow * Cx + kc;
        Frag16 a;
        a.u[0] = *(const u32x4*)(xr + half * 8);
        a.u[1] = *(const u32x4*)(xr + 16 + half * 8);
        #pragma unroll
        for (int m = 0; m < 3; ++m) {
            const unsigned short* WT = wT + m * (Hx * Cx);
            Frag16 bfr[4];
            #pragma unroll
            for (int nt = 0; nt < 4; ++nt) {
                const unsigned short* p = WT + (nt * 16 + lo) * Cx + kc + half * 8;
                bfr[nt].u[0] = *(const u32x4*)p;
                bfr[nt].u[1] = *(const u32x4*)(p + 16);
            }
            #pragma unroll
            for (int nt = 0; nt < 4; ++nt)
                acc[m][nt] = __builtin_amdgcn_wmma_f32_16x16x32_bf16(
                        false, a.v, false, bfr[nt].v, (short)0, acc[m][nt], false, false);
        }
    }

    // store: C/D layout row m = r + 8*half, col n = nt*16 + lo
    const int b = rowbase >> 11;
    const int jl0 = (rowbase & (Tx - 1)) + 8 * half;   // rows r=0..7 -> jl0+r
    #pragma unroll
    for (int nt = 0; nt < 4; ++nt) {
        const int col = nt * 16 + lo;
        unsigned short vpack[8];
        #pragma unroll
        for (int r = 0; r < 8; ++r) {
            const int row = rowbase + r + 8 * half;
            qb[row * Hx + col] = f2bf_bits(acc[0][nt][r]);
            kb[row * Hx + col] = f2bf_bits(acc[1][nt][r]);
            vpack[r] = f2bf_bits(acc[2][nt][r]);
        }
        // transposed V: consecutive r -> consecutive j -> one 16B store
        *(u32x4*)(vtb + ((size_t)b * Hx + col) * Tx + jl0) = *(const u32x4*)vpack;
    }
}

// ---------------------------------------------------------------------------
// Kernel 2 (fallback, used when d_ws can't hold bf16 x): reads fp32 x and
// converts A fragments on the fly.
// ---------------------------------------------------------------------------
__global__ __launch_bounds__(128) void qkv_proj_f32(
        const float* __restrict__ x,
        const unsigned short* __restrict__ wT,
        unsigned short* __restrict__ qb,
        unsigned short* __restrict__ kb,
        unsigned short* __restrict__ vtb) {
    const int lane = threadIdx.x & 31;
    const int lo   = lane & 15;
    const int half = lane >> 4;
    const int wid  = blockIdx.x * (blockDim.x >> 5) + (threadIdx.x >> 5);
    const int rowbase = wid << 4;
    const int arow = rowbase + lo;

    const v8f vz = {0.f, 0.f, 0.f, 0.f, 0.f, 0.f, 0.f, 0.f};
    v8f acc[3][4];
    #pragma unroll
    for (int m = 0; m < 3; ++m)
        #pragma unroll
        for (int nt = 0; nt < 4; ++nt) acc[m][nt] = vz;

    for (int kc = 0; kc < Cx; kc += 32) {
        const float* xr = x + (size_t)arow * Cx + kc;
        f32x4 f0 = *(const f32x4*)(xr + half * 8);
        f32x4 f1 = *(const f32x4*)(xr + half * 8 + 4);
        f32x4 f2 = *(const f32x4*)(xr + 16 + half * 8);
        f32x4 f3 = *(const f32x4*)(xr + 16 + half * 8 + 4);
        Frag16 a;
        #pragma unroll
        for (int e = 0; e < 4; ++e) {
            a.v[e]      = __builtin_bit_cast(__bf16, f2bf_bits(f0[e]));
            a.v[4 + e]  = __builtin_bit_cast(__bf16, f2bf_bits(f1[e]));
            a.v[8 + e]  = __builtin_bit_cast(__bf16, f2bf_bits(f2[e]));
            a.v[12 + e] = __builtin_bit_cast(__bf16, f2bf_bits(f3[e]));
        }
        #pragma unroll
        for (int m = 0; m < 3; ++m) {
            const unsigned short* WT = wT + m * (Hx * Cx);
            Frag16 bfr[4];
            #pragma unroll
            for (int nt = 0; nt < 4; ++nt) {
                const unsigned short* p = WT + (nt * 16 + lo) * Cx + kc + half * 8;
                bfr[nt].u[0] = *(const u32x4*)p;
                bfr[nt].u[1] = *(const u32x4*)(p + 16);
            }
            #pragma unroll
            for (int nt = 0; nt < 4; ++nt)
                acc[m][nt] = __builtin_amdgcn_wmma_f32_16x16x32_bf16(
                        false, a.v, false, bfr[nt].v, (short)0, acc[m][nt], false, false);
        }
    }

    const int b = rowbase >> 11;
    const int jl0 = (rowbase & (Tx - 1)) + 8 * half;
    #pragma unroll
    for (int nt = 0; nt < 4; ++nt) {
        const int col = nt * 16 + lo;
        unsigned short vpack[8];
        #pragma unroll
        for (int r = 0; r < 8; ++r) {
            const int row = rowbase + r + 8 * half;
            qb[row * Hx + col] = f2bf_bits(acc[0][nt][r]);
            kb[row * Hx + col] = f2bf_bits(acc[1][nt][r]);
            vpack[r] = f2bf_bits(acc[2][nt][r]);
        }
        *(u32x4*)(vtb + ((size_t)b * Hx + col) * Tx + jl0) = *(const u32x4*)vpack;
    }
}

// ---------------------------------------------------------------------------
// Kernel 3: flash attention, async-DMA double-buffered LDS staging.
// Block = 4 waves = 4 consecutive i-tiles of one batch (uniform j loop).
// Pipeline per 32-wide j-chunk:
//   s_wait_asynccnt 0  -> own async copies for buf[cur] have landed
//   s_barrier          -> all waves' copies for buf[cur] published
//   issue async copies for buf[next]  (overlaps with compute below)
//   compute: S = K.Q^T (4 WMMA), causal mask, base-2 online softmax,
//            P->LDS->A-layout, O += P @ V (4 WMMA)
// ---------------------------------------------------------------------------
__global__ __launch_bounds__(128) void attn(
        const unsigned short* __restrict__ qb,
        const unsigned short* __restrict__ kbm,
        const unsigned short* __restrict__ vtb,
        float* __restrict__ out) {
    __shared__ unsigned short qs2[2][32 * QS_STRIDE];   // [j_local][d], padded
    __shared__ unsigned short vs2[2][64 * VS_STRIDE];   // [d][j_local], padded
    __shared__ unsigned short pbuf[4][16 * PB_STRIDE];  // per-wave P tile

    const int tid  = threadIdx.x;
    const int lane = tid & 31;
    const int warp = tid >> 5;
    const int lo   = lane & 15;
    const int half = lane >> 4;

    const int bid   = blockIdx.x;          // 0..255
    const int b     = bid >> 5;            // batch
    const int group = bid & 31;            // 4-tile group within batch
    const int itile = group * 4 + warp;
    const int ibase = itile << 4;
    const int jmax  = (group + 1) * 64;    // uniform per block
    const float k2 = 0.03125f * 1.44269504088896340736f;  // log2e / sqrt(C)

    const unsigned short* qgb = qb + (size_t)b * Tx * Hx;   // [T][64]
    const unsigned short* vgb = vtb + (size_t)b * Hx * Tx;  // [64][T]

    // thread's staging share: 2x q units (32 rows x 8 units) + 2x v units
    const int qu0 = tid,        qu1 = tid + 128;
    const int vu0 = tid,        vu1 = tid + 128;

    // K A-fragments (rows ibase..ibase+15, d=0..63): hoisted, reused all j
    Frag16 ak[2];
    const unsigned short* kr = kbm + ((size_t)(b * Tx + ibase + lo)) * Hx;
    #pragma unroll
    for (int dc = 0; dc < 2; ++dc) {
        ak[dc].u[0] = *(const u32x4*)(kr + dc * 32 + half * 8);
        ak[dc].u[1] = *(const u32x4*)(kr + dc * 32 + 16 + half * 8);
    }

    const v8f vz = {0.f, 0.f, 0.f, 0.f, 0.f, 0.f, 0.f, 0.f};
    v8f o[4];
    #pragma unroll
    for (int nt = 0; nt < 4; ++nt) o[nt] = vz;
    float mrow[8], lsum[8];
    #pragma unroll
    for (int r = 0; r < 8; ++r) { mrow[r] = -__builtin_inff(); lsum[r] = 0.f; }

    // ---- prologue: async-stage chunk 0 into buffer 0
    {
        int jr, c8, dr, j8;
        jr = qu0 >> 3; c8 = (qu0 & 7) * 8;
        async_copy16(&qs2[0][jr * QS_STRIDE + c8], qgb + (size_t)jr * Hx + c8);
        jr = qu1 >> 3; c8 = (qu1 & 7) * 8;
        async_copy16(&qs2[0][jr * QS_STRIDE + c8], qgb + (size_t)jr * Hx + c8);
        dr = vu0 >> 2; j8 = (vu0 & 3) * 8;
        async_copy16(&vs2[0][dr * VS_STRIDE + j8], vgb + (size_t)dr * Tx + j8);
        dr = vu1 >> 2; j8 = (vu1 & 3) * 8;
        async_copy16(&vs2[0][dr * VS_STRIDE + j8], vgb + (size_t)dr * Tx + j8);
    }

    for (int jb = 0; jb < jmax; jb += 32) {
        const int sel = (jb >> 5) & 1;
        wait_asynccnt0();          // own DMA for buf[sel] landed
        __syncthreads();           // everyone's DMA for buf[sel] published

        // ---- async prefetch of next chunk into buf[1-sel] (uniform cond)
        if (jb + 32 < jmax) {
            const int jn = jb + 32;
            int jr, c8, dr, j8;
            jr = qu0 >> 3; c8 = (qu0 & 7) * 8;
            async_copy16(&qs2[1 - sel][jr * QS_STRIDE + c8],
                         qgb + (size_t)(jn + jr) * Hx + c8);
            jr = qu1 >> 3; c8 = (qu1 & 7) * 8;
            async_copy16(&qs2[1 - sel][jr * QS_STRIDE + c8],
                         qgb + (size_t)(jn + jr) * Hx + c8);
            dr = vu0 >> 2; j8 = (vu0 & 3) * 8;
            async_copy16(&vs2[1 - sel][dr * VS_STRIDE + j8],
                         vgb + (size_t)dr * Tx + jn + j8);
            dr = vu1 >> 2; j8 = (vu1 & 3) * 8;
            async_copy16(&vs2[1 - sel][dr * VS_STRIDE + j8],
                         vgb + (size_t)dr * Tx + jn + j8);
        }

        if (jb < ibase + 16) {             // wave still below its diagonal
            const unsigned short* qs = qs2[sel];
            const unsigned short* vs = vs2[sel];

            // ---- S tiles from LDS: two 16x16 column tiles, K=64 in 2 WMMAs
            v8f s[2];
            #pragma unroll
            for (int t = 0; t < 2; ++t) {
                const unsigned short* qrow = qs + (t * 16 + lo) * QS_STRIDE;
                Frag16 bq0, bq1;
                bq0.u[0] = *(const u32x4*)(qrow + half * 8);
                bq0.u[1] = *(const u32x4*)(qrow + half * 8 + 16);
                bq1.u[0] = *(const u32x4*)(qrow + 32 + half * 8);
                bq1.u[1] = *(const u32x4*)(qrow + 32 + half * 8 + 16);
                v8f accs = __builtin_amdgcn_wmma_f32_16x16x32_bf16(
                        false, ak[0].v, false, bq0.v, (short)0, vz, false, false);
                accs = __builtin_amdgcn_wmma_f32_16x16x32_bf16(
                        false, ak[1].v, false, bq1.v, (short)0, accs, false, false);
                s[t] = accs;
            }

            // ---- scale + causal mask + base-2 online softmax (m = r+8*half)
            float corr[8], p0[8], p1[8];
            #pragma unroll
            for (int r = 0; r < 8; ++r) {
                const int ig = ibase + r + 8 * half;   // global row i
                float s0 = s[0][r] * k2;
                float s1 = s[1][r] * k2;
                if (jb + lo > ig)      s0 = -__builtin_inff();
                if (jb + 16 + lo > ig) s1 = -__builtin_inff();
                float v = fmaxf(s0, s1);
                v = fmaxf(v, __shfl_xor(v, 8, 32));
                v = fmaxf(v, __shfl_xor(v, 4, 32));
                v = fmaxf(v, __shfl_xor(v, 2, 32));
                v = fmaxf(v, __shfl_xor(v, 1, 32));
                const float mn = fmaxf(mrow[r], v);
                const float c  = __builtin_exp2f(mrow[r] - mn);   // v_exp_f32
                mrow[r] = mn;
                const float e0 = __builtin_exp2f(s0 - mn);
                const float e1 = __builtin_exp2f(s1 - mn);
                float sm = e0 + e1;
                sm += __shfl_xor(sm, 8, 32);
                sm += __shfl_xor(sm, 4, 32);
                sm += __shfl_xor(sm, 2, 32);
                sm += __shfl_xor(sm, 1, 32);
                lsum[r] = lsum[r] * c + sm;
                corr[r] = c;
                p0[r] = e0;
                p1[r] = e1;
            }

            // ---- rescale O, stage P (C layout) into wave-private LDS
            #pragma unroll
            for (int r = 0; r < 8; ++r) {
                o[0][r] *= corr[r];
                o[1][r] *= corr[r];
                o[2][r] *= corr[r];
                o[3][r] *= corr[r];
                const int m = r + 8 * half;
                pbuf[warp][m * PB_STRIDE + lo]      = f2bf_bits(p0[r]);
                pbuf[warp][m * PB_STRIDE + 16 + lo] = f2bf_bits(p1[r]);
            }
            __builtin_amdgcn_wave_barrier();  // wave-sync; LDS in-order/wave

            // ---- reload P in A layout: two ds_load_b128
            Frag16 ap;
            const unsigned short* pb = &pbuf[warp][lo * PB_STRIDE];
            ap.u[0] = *(const u32x4*)(pb + half * 8);
            ap.u[1] = *(const u32x4*)(pb + 16 + half * 8);
            __builtin_amdgcn_wave_barrier();

            // ---- O += P @ V from staged LDS
            #pragma unroll
            for (int nt = 0; nt < 4; ++nt) {
                const unsigned short* vrow = vs + (nt * 16 + lo) * VS_STRIDE;
                Frag16 bv;
                bv.u[0] = *(const u32x4*)(vrow + half * 8);
                bv.u[1] = *(const u32x4*)(vrow + half * 8 + 16);
                o[nt] = __builtin_amdgcn_wmma_f32_16x16x32_bf16(
                        false, ap.v, false, bv.v, (short)0, o[nt], false, false);
            }
        }
    }

    // ---- normalize and store (fp32 output)
    #pragma unroll
    for (int r = 0; r < 8; ++r) {
        const float inv = 1.0f / lsum[r];
        const int row = ibase + r + 8 * half;
        float* orow = out + ((size_t)(b * Tx + row)) * Hx;
        orow[lo]      = o[0][r] * inv;
        orow[16 + lo] = o[1][r] * inv;
        orow[32 + lo] = o[2][r] * inv;
        orow[48 + lo] = o[3][r] * inv;
    }
}

// ---------------------------------------------------------------------------
// Host launcher.  d_in order: x, Wq, Wk, Wv.  d_out: [B,T,H] fp32.
// Workspace layout (bf16 elements):
//   wT  : 3*64*1024          qb : B*T*64        kb : B*T*64
//   vtb : B*64*T             xb : B*T*C   (only if ws_size permits)
// ---------------------------------------------------------------------------
extern "C" void kernel_launch(void* const* d_in, const int* in_sizes, int n_in,
                              void* d_out, int out_size, void* d_ws, size_t ws_size,
                              hipStream_t stream) {
    const float* x  = (const float*)d_in[0];
    const float* Wq = (const float*)d_in[1];
    const float* Wk = (const float*)d_in[2];
    const float* Wv = (const float*)d_in[3];

    unsigned short* wT  = (unsigned short*)d_ws;
    unsigned short* qb  = wT + 3 * Hx * Cx;
    unsigned short* kb  = qb + (size_t)Bx * Tx * Hx;
    unsigned short* vtb = kb + (size_t)Bx * Tx * Hx;
    unsigned short* xb  = vtb + (size_t)Bx * Tx * Hx;

    const size_t base_elems = (size_t)3 * Hx * Cx + (size_t)3 * Bx * Tx * Hx;
    const size_t xb_elems   = (size_t)Bx * Tx * Cx;
    const bool   have_xb    = ws_size >= (base_elems + xb_elems) * sizeof(unsigned short);

    prep_w<<<(3 * Hx * Cx + 255) / 256, 256, 0, stream>>>(Wq, Wk, Wv, wT);

    if (have_xb) {
        prep_x<<<(int)(xb_elems / (256 * 8)), 256, 0, stream>>>(x, xb);
        qkv_proj_bf16<<<256, 128, 0, stream>>>(xb, wT, qb, kb, vtb);
    } else {
        qkv_proj_f32<<<256, 128, 0, stream>>>(x, wT, qb, kb, vtb);
    }

    // 8 batches x 32 groups = 256 blocks; 4 waves/block = 1024 i-tiles
    attn<<<256, 128, 0, stream>>>(qb, kb, vtb, (float*)d_out);
}
